// NeighborWindowAttention_317827580592
// MI455X (gfx1250) — matrix operand order repaired and checked by reading, not compiled, not run on hardware
//
#include <hip/hip_runtime.h>
#include <hip/hip_bf16.h>

typedef __attribute__((ext_vector_type(16))) _Float16 v16h;
typedef __attribute__((ext_vector_type(8)))  _Float16 v8h;
typedef __attribute__((ext_vector_type(8)))  float    v8f;

// A-fragment (16xK rows striped in-lane): per lane two 16B chunks at +0 and
// +8 halves; caller adds row*stride + (lane>>4)*16.
__device__ __forceinline__ v16h frag_ld_a(const _Float16* p) {
    v8h lo = *(const v8h*)(p);
    v8h hi = *(const v8h*)(p + 8);
    return __builtin_shufflevector(lo, hi, 0,1,2,3,4,5,6,7,8,9,10,11,12,13,14,15);
}

// B-fragment (Kx16, lane = column): 16 contiguous K-values per lane-half;
// caller adds col*stride + (lane>>4)*16. One contiguous 32B read.
__device__ __forceinline__ v16h frag_ld_b(const _Float16* p) {
    v8h lo = *(const v8h*)(p);
    v8h hi = *(const v8h*)(p + 8);
    return __builtin_shufflevector(lo, hi, 0,1,2,3,4,5,6,7,8,9,10,11,12,13,14,15);
}

__device__ __forceinline__ v8f wmma_f16(v16h a, v16h b, v8f c) {
    return __builtin_amdgcn_wmma_f32_16x16x32_f16(false, a, false, b, (short)0, c, false, false);
}

// ---------------------------------------------------------------------------
// Weight conversion f32 -> f16
// ---------------------------------------------------------------------------
__global__ void cvt_f2h(const float* __restrict__ src, _Float16* __restrict__ dst, int n) {
    int i = blockIdx.x * blockDim.x + threadIdx.x;
    if (i < n) dst[i] = (_Float16)src[i];
}

// ---------------------------------------------------------------------------
// Transposed relative-position bias rbT[h][j][q] (8 x 576 x 64 f32)
// ---------------------------------------------------------------------------
__global__ void prep_biasT(const float* __restrict__ bt, float* __restrict__ rbT) {
    int i = blockIdx.x * blockDim.x + threadIdx.x;
    if (i >= 8 * 576 * 64) return;
    int q = i & 63;
    int t = i >> 6;
    int j = t % 576;
    int h = t / 576;
    int jr = j / 24, jc = j % 24;                     // key coords in 24x24
    int idx = ((q >> 3) - jr + 23) * 31 + ((q & 7) - jc + 23);
    rbT[i] = bt[idx * 8 + h];
}

// ---------------------------------------------------------------------------
// Transposed window mask mT[w][j][q] (64 x 576 x 64 f32)
// ---------------------------------------------------------------------------
__global__ void prep_maskT(const float* __restrict__ mask, float* __restrict__ mT) {
    int i = blockIdx.x * blockDim.x + threadIdx.x;
    if (i >= 64 * 576 * 64) return;
    int q = i & 63;
    int t = i >> 6;
    int j = t % 576;
    int w = t / 576;
    mT[i] = mask[(w * 64 + q) * 576 + j];
}

// ---------------------------------------------------------------------------
// Y = X @ W^T + bias, K = N = 256 fixed. 64x64 tile per workgroup, 8 waves.
// MODE 0: Y f16 row-major [M][256]
// MODE 1: Y f16 transposed per head: Y[((b*8+h)*32+d)*576 + j], X rows = b*576+j
// MODE 2: Y f32 row-major [M][256]
// ---------------------------------------------------------------------------
template <typename TIN, int MODE>
__global__ __launch_bounds__(256)
void gemm_xwT(const TIN* __restrict__ X, const _Float16* __restrict__ W16,
              const float* __restrict__ bias, void* __restrict__ Yv)
{
    constexpr int LP = 40;                       // LDS row pitch (halves), 80B
    __shared__ __align__(16) _Float16 Xs[64 * LP];
    __shared__ __align__(16) _Float16 Ws[64 * LP];

    const int tid  = threadIdx.x;
    const int lane = tid & 31;
    const int wid  = tid >> 5;
    const int l16  = lane & 15;
    const int hi   = lane >> 4;
    const int wm   = wid & 3;                    // 4 m-subtiles of 16
    const int wn   = wid >> 2;                   // 2 n-subtiles of 32

    const long m0 = (long)blockIdx.x * 64;
    const int  n0 = blockIdx.y * 64;

    const int lr = tid >> 2;                     // 0..63 (tile row)
    const int ls = (tid & 3) * 8;                // 0,8,16,24 (k-chunk)

    v8f acc0 = {}, acc1 = {};

    for (int kc = 0; kc < 256; kc += 32) {
        __syncthreads();
        if constexpr (sizeof(TIN) == 4) {
            const float* xp = (const float*)X + (m0 + lr) * 256 + kc + ls;
            float4 a = *(const float4*)xp;
            float4 c = *(const float4*)(xp + 4);
            v8h d;
            d[0]=(_Float16)a.x; d[1]=(_Float16)a.y; d[2]=(_Float16)a.z; d[3]=(_Float16)a.w;
            d[4]=(_Float16)c.x; d[5]=(_Float16)c.y; d[6]=(_Float16)c.z; d[7]=(_Float16)c.w;
            *(v8h*)(Xs + lr * LP + ls) = d;
            if (kc < 224) __builtin_prefetch((const void*)(xp + 32), 0, 3);
        } else {
            const _Float16* xp = (const _Float16*)X + (m0 + lr) * 256 + kc + ls;
            *(v8h*)(Xs + lr * LP + ls) = *(const v8h*)xp;
            if (kc < 224) __builtin_prefetch((const void*)(xp + 32), 0, 3);
        }
        {
            const _Float16* wp = W16 + (long)(n0 + lr) * 256 + kc + ls;
            *(v8h*)(Ws + lr * LP + ls) = *(const v8h*)wp;
            if (kc < 224) __builtin_prefetch((const void*)(wp + 32), 0, 3);
        }
        __syncthreads();

        v16h a  = frag_ld_a(Xs + (wm * 16 + l16) * LP + hi * 16);
        v16h b0 = frag_ld_b(Ws + (wn * 32 + l16) * LP + hi * 16);
        v16h b1 = frag_ld_b(Ws + (wn * 32 + 16 + l16) * LP + hi * 16);
        acc0 = wmma_f16(a, b0, acc0);
        acc1 = wmma_f16(a, b1, acc1);
    }

    const int ncol0 = n0 + wn * 32;
    const float bb0 = bias[ncol0 + l16];
    const float bb1 = bias[ncol0 + 16 + l16];

    if constexpr (MODE == 0) {
        _Float16* Y = (_Float16*)Yv;
        #pragma unroll
        for (int r = 0; r < 8; ++r) {
            long m = m0 + wm * 16 + hi * 8 + r;
            Y[m * 256 + ncol0 + l16]      = (_Float16)(acc0[r] + bb0);
            Y[m * 256 + ncol0 + 16 + l16] = (_Float16)(acc1[r] + bb1);
        }
    } else if constexpr (MODE == 2) {
        float* Y = (float*)Yv;
        #pragma unroll
        for (int r = 0; r < 8; ++r) {
            long m = m0 + wm * 16 + hi * 8 + r;
            Y[m * 256 + ncol0 + l16]      = acc0[r] + bb0;
            Y[m * 256 + ncol0 + 16 + l16] = acc1[r] + bb1;
        }
    } else {
        // Transposed per-head store: lane holds fixed output column and 8
        // consecutive X rows -> one contiguous 16B store per fragment.
        _Float16* Y = (_Float16*)Yv;
        long mrow = m0 + wm * 16 + hi * 8;       // 8-row aligned, within one b
        int b = (int)(mrow / 576);
        int j = (int)(mrow % 576);
        v8h t0, t1;
        #pragma unroll
        for (int r = 0; r < 8; ++r) {
            t0[r] = (_Float16)(acc0[r] + bb0);
            t1[r] = (_Float16)(acc1[r] + bb1);
        }
        int ng0 = ncol0 + l16, ng1 = ncol0 + 16 + l16;
        long row0 = (long)((b * 8 + (ng0 >> 5)) * 32 + (ng0 & 31));
        long row1 = (long)((b * 8 + (ng1 >> 5)) * 32 + (ng1 & 31));
        *(v8h*)(Y + row0 * 576 + j) = t0;
        *(v8h*)(Y + row1 * 576 + j) = t1;
    }
}

// ---------------------------------------------------------------------------
// Attention: one 4-wave block per (window b, head h); wave w owns q rows
// w*16..w*16+15. K (576x32) and Vt (32x576) head slices are staged once into
// LDS and shared by all 4 waves. Scores go to a per-wave f16 LDS buffer with
// transposed bias/mask added via float4 loads; softmax rewrites exp in place;
// O = P V from LDS; normalize at the epilogue. LDS ~150KB -> 2 blocks/WGP.
// ---------------------------------------------------------------------------
#define SP 592   // score pitch in halves (576 + 16), 16B multiple

__global__ __launch_bounds__(128)
void attn_kernel(const _Float16* __restrict__ q16, const _Float16* __restrict__ k16,
                 const _Float16* __restrict__ vt16, const float* __restrict__ mT,
                 const float* __restrict__ rbT, _Float16* __restrict__ xh16)
{
    __shared__ __align__(16) _Float16 Ks[576 * 32];     // 36 KB
    __shared__ __align__(16) _Float16 Vs[32 * 576];     // 36 KB
    __shared__ __align__(16) _Float16 Sbuf[4 * 16 * SP];// 74 KB (per-wave slices)
    __shared__ float sums[64];

    const int tid  = threadIdx.x;
    const int lane = tid & 31;
    const int wid  = tid >> 5;
    const int l16  = lane & 15;
    const int hi   = lane >> 4;
    const int b    = blockIdx.x;
    const int h    = blockIdx.y;
    const int q0   = wid * 16;                   // this wave's query-row base

    const float scale = 0.17677669529663687f;    // 32^-0.5

    // ---- cooperative staging of K and Vt head slices
    for (int t = tid; t < 576 * 4; t += 128) {
        int row = t >> 2, c = (t & 3) * 8;
        *(v8h*)(Ks + row * 32 + c) =
            *(const v8h*)(k16 + (long)(b * 576 + row) * 256 + h * 32 + c);
    }
    const _Float16* vtb = vt16 + (long)((b * 8 + h) * 32) * 576;
    for (int t = tid; t < 2304; t += 128) {
        *(v8h*)(Vs + t * 8) = *(const v8h*)(vtb + t * 8);
    }

    // Q fragment (A, 16x32), scale folded in (overlaps with staging loads)
    v16h aq = frag_ld_a(q16 + ((long)(b * 64 + q0 + l16) * 256 + h * 32 + hi * 16));
    #pragma unroll
    for (int e = 0; e < 16; ++e) aq[e] = (_Float16)((float)aq[e] * scale);

    __syncthreads();

    _Float16* Sw = Sbuf + wid * 16 * SP;
    float*    sw = sums + wid * 16;

    // ---- scores: 36 WMMAs over 576 keys, B-fragments from LDS
    for (int jt = 0; jt < 36; ++jt) {
        const int j0 = jt * 16;
        v16h bk = frag_ld_b(Ks + (j0 + l16) * 32 + hi * 16);
        v8f s = {};
        s = wmma_f16(aq, bk, s);
        const int j = j0 + l16;
        // transposed tables: 8 consecutive q values per lane -> 2x float4 each
        float rbv[8], mv[8];
        {
            const float4* rp = (const float4*)(rbT + ((long)h * 576 + j) * 64 + q0 + hi * 8);
            const float4* mp = (const float4*)(mT + ((long)(b & 63) * 576 + j) * 64 + q0 + hi * 8);
            *(float4*)&rbv[0] = rp[0];
            *(float4*)&rbv[4] = rp[1];
            *(float4*)&mv[0]  = mp[0];
            *(float4*)&mv[4]  = mp[1];
        }
        #pragma unroll
        for (int r = 0; r < 8; ++r) {
            int qloc = r + hi * 8;               // 0..15 local row
            Sw[qloc * SP + j] = (_Float16)(s[r] + rbv[r] + mv[r]);
        }
    }
    asm volatile("s_wait_dscnt 0" ::: "memory");

    // ---- wave-local softmax per row, exp written in place (unnormalized)
    for (int r = 0; r < 16; ++r) {
        float m = -1e30f;
        for (int j = lane; j < 576; j += 32) m = fmaxf(m, (float)Sw[r * SP + j]);
        #pragma unroll
        for (int off = 16; off > 0; off >>= 1) m = fmaxf(m, __shfl_xor(m, off, 32));
        float ss = 0.f;
        for (int j = lane; j < 576; j += 32) {
            float e = __expf((float)Sw[r * SP + j] - m);
            ss += e;
            Sw[r * SP + j] = (_Float16)e;
        }
        #pragma unroll
        for (int off = 16; off > 0; off >>= 1) ss += __shfl_xor(ss, off, 32);
        if (lane == 0) sw[r] = ss;
    }
    asm volatile("s_wait_dscnt 0" ::: "memory");

    // ---- O = P V : 18 k-steps x 2 d-subtiles = 36 WMMAs, all operands in LDS
    v8f o0 = {}, o1 = {};
    for (int jt = 0; jt < 18; ++jt) {
        const int j0 = jt * 32;
        v16h ap = frag_ld_a(Sw + l16 * SP + j0 + hi * 16);
        v16h b0 = frag_ld_b(Vs + (l16)      * 576 + j0 + hi * 16);
        v16h b1 = frag_ld_b(Vs + (16 + l16) * 576 + j0 + hi * 16);
        o0 = wmma_f16(ap, b0, o0);
        o1 = wmma_f16(ap, b1, o1);
    }

    #pragma unroll
    for (int r = 0; r < 8; ++r) {
        int qloc  = r + hi * 8;
        float inv = 1.0f / sw[qloc];
        long row  = (long)(b * 64 + q0 + qloc) * 256 + h * 32;
        xh16[row + l16]      = (_Float16)(o0[r] * inv);
        xh16[row + 16 + l16] = (_Float16)(o1[r] * inv);
    }
}

// ---------------------------------------------------------------------------
extern "C" void kernel_launch(void* const* d_in, const int* in_sizes, int n_in,
                              void* d_out, int out_size, void* d_ws, size_t ws_size,
                              hipStream_t stream) {
    const float* q      = (const float*)d_in[0];
    const float* k      = (const float*)d_in[1];
    const float* v      = (const float*)d_in[2];
    const float* mask   = (const float*)d_in[3];
    const float* Wq_w   = (const float*)d_in[4];
    const float* Wq_b   = (const float*)d_in[5];
    const float* Wk_w   = (const float*)d_in[6];
    const float* Wk_b   = (const float*)d_in[7];
    const float* Wv_w   = (const float*)d_in[8];
    const float* Wv_b   = (const float*)d_in[9];
    const float* btab   = (const float*)d_in[10];
    const float* proj_w = (const float*)d_in[11];
    const float* proj_b = (const float*)d_in[12];

    // workspace layout (halves)
    _Float16* ws   = (_Float16*)d_ws;
    _Float16* Wq16 = ws;
    _Float16* Wk16 = Wq16 + 65536;
    _Float16* Wv16 = Wk16 + 65536;
    _Float16* Wp16 = Wv16 + 65536;
    _Float16* q16  = Wp16 + 65536;                       // [65536][256]
    _Float16* k16  = q16  + (long)65536  * 256;          // [589824][256]
    _Float16* vt16 = k16  + (long)589824 * 256;          // [1024*8*32][576]
    _Float16* xh16 = vt16 + (long)589824 * 256;          // [65536][256]
    float*    rbT  = (float*)(xh16 + (long)65536 * 256); // [8][576][64]
    float*    mT   = rbT + (long)8 * 576 * 64;           // [64][576][64]

    cvt_f2h<<<256, 256, 0, stream>>>(Wq_w,   Wq16, 65536);
    cvt_f2h<<<256, 256, 0, stream>>>(Wk_w,   Wk16, 65536);
    cvt_f2h<<<256, 256, 0, stream>>>(Wv_w,   Wv16, 65536);
    cvt_f2h<<<256, 256, 0, stream>>>(proj_w, Wp16, 65536);
    prep_biasT<<<1152, 256, 0, stream>>>(btab, rbT);
    prep_maskT<<<9216, 256, 0, stream>>>(mask, mT);

    gemm_xwT<float, 0><<<dim3(1024, 4), 256, 0, stream>>>(q, Wq16, Wq_b, (void*)q16);
    gemm_xwT<float, 0><<<dim3(9216, 4), 256, 0, stream>>>(k, Wk16, Wk_b, (void*)k16);
    gemm_xwT<float, 1><<<dim3(9216, 4), 256, 0, stream>>>(v, Wv16, Wv_b, (void*)vt16);

    attn_kernel<<<dim3(1024, 8), 128, 0, stream>>>(q16, k16, vt16, mT, rbT, xh16);

    gemm_xwT<_Float16, 2><<<dim3(1024, 4), 256, 0, stream>>>(xh16, Wp16, proj_b, d_out);
}
